// CharRNN_15513421873236
// MI455X (gfx1250) — matrix operand order repaired and checked by reading, not compiled, not run on hardware
//
#include <hip/hip_runtime.h>
#include <hip/hip_bf16.h>

#define SEQ   65536
#define EMB   512
#define HID   1024
#define NCH   512

typedef __attribute__((ext_vector_type(16))) __bf16 v16bf;
typedef __attribute__((ext_vector_type(8)))  float  v8f;
typedef __attribute__((ext_vector_type(4)))  int    i4x;

union BF16Tile {
    v16bf v;
    i4x   q[2];
};

__device__ __forceinline__ v8f wmma_bf16(const BF16Tile& a, const BF16Tile& b, v8f c) {
    return __builtin_amdgcn_wmma_f32_16x16x32_bf16(false, a.v, false, b.v,
                                                   (short)0, c, false, false);
}

__device__ __forceinline__ unsigned short f32_to_bf16(float f) {
    unsigned int x = __float_as_uint(f);
    x += 0x7FFFu + ((x >> 16) & 1u);          // round-to-nearest-even
    return (unsigned short)(x >> 16);
}

// ---------------- fp32 -> bf16 weight conversion ----------------
__global__ void k_cvt_bf16(const float* __restrict__ src,
                           unsigned short* __restrict__ dst, int n) {
    int i = blockIdx.x * blockDim.x + threadIdx.x;
    if (i < n) dst[i] = f32_to_bf16(src[i]);
}

// ---------------- embedding gather + bf16 convert ----------------
__global__ void k_embed(const int* __restrict__ seq,
                        const float* __restrict__ emb,
                        unsigned short* __restrict__ out) {
    int i = blockIdx.x * blockDim.x + threadIdx.x;   // over SEQ*EMB
    int s = i >> 9;
    int e = i & (EMB - 1);
    out[i] = f32_to_bf16(emb[(size_t)seq[s] * EMB + e]);
}

// ---------------- x_proj = emb @ Wax^T  (WMMA bf16, 2x2 register block) ----------------
// A = emb_bf16 [SEQ, EMB] row-major, B[k,n] = Wax[n, k] (Wax row-major [HID,EMB])
__global__ void __launch_bounds__(128)
k_xproj(const unsigned short* __restrict__ A,
        const unsigned short* __restrict__ B,
        float* __restrict__ C) {
    const int lane = threadIdx.x & 31;
    const int wave = threadIdx.x >> 5;         // 0..3
    const int half = lane >> 4;                // 0/1
    const int l16  = lane & 15;
    const int sBase = blockIdx.x * 32;
    const int hBase = (blockIdx.y * 4 + wave) * 32;
    const unsigned short* arow0 = A + (size_t)(sBase + l16) * EMB;
    const unsigned short* arow1 = arow0 + (size_t)16 * EMB;
    const unsigned short* brow0 = B + (size_t)(hBase + l16) * EMB;
    const unsigned short* brow1 = brow0 + (size_t)16 * EMB;
    v8f acc00 = {0.f,0.f,0.f,0.f,0.f,0.f,0.f,0.f};
    v8f acc01 = acc00, acc10 = acc00, acc11 = acc00;
#pragma unroll 2
    for (int k = 0; k < EMB; k += 32) {
        BF16Tile a0, a1, b0, b1;
        const int aoff = k + 16 * half;        // A 16x32: one contiguous 16-elem run
        const int boff = k + 16 * half;        // B 32x16: one contiguous 16-elem run
        a0.q[0] = *reinterpret_cast<const i4x*>(arow0 + aoff);
        a0.q[1] = *reinterpret_cast<const i4x*>(arow0 + aoff + 8);
        a1.q[0] = *reinterpret_cast<const i4x*>(arow1 + aoff);
        a1.q[1] = *reinterpret_cast<const i4x*>(arow1 + aoff + 8);
        b0.q[0] = *reinterpret_cast<const i4x*>(brow0 + boff);
        b0.q[1] = *reinterpret_cast<const i4x*>(brow0 + boff + 8);
        b1.q[0] = *reinterpret_cast<const i4x*>(brow1 + boff);
        b1.q[1] = *reinterpret_cast<const i4x*>(brow1 + boff + 8);
        acc00 = wmma_bf16(a0, b0, acc00);
        acc01 = wmma_bf16(a0, b1, acc01);
        acc10 = wmma_bf16(a1, b0, acc10);
        acc11 = wmma_bf16(a1, b1, acc11);
    }
#pragma unroll
    for (int j = 0; j < 8; ++j) {
        const int r0 = sBase + j + 8 * half;
        const int r1 = r0 + 16;
        C[(size_t)r0 * HID + hBase + l16]      = acc00[j];
        C[(size_t)r0 * HID + hBase + 16 + l16] = acc01[j];
        C[(size_t)r1 * HID + hBase + l16]      = acc10[j];
        C[(size_t)r1 * HID + hBase + 16 + l16] = acc11[j];
    }
}

// ---------------- out = hs @ Wya^T + b_y  (WMMA bf16, 2x2 register block) ----------------
__global__ void __launch_bounds__(128)
k_out(const unsigned short* __restrict__ A,   // hs_bf16 [SEQ, HID]
      const unsigned short* __restrict__ B,   // Wya_bf16 [NCH, HID]
      const float* __restrict__ bias,
      float* __restrict__ C) {                // [SEQ, NCH]
    const int lane = threadIdx.x & 31;
    const int wave = threadIdx.x >> 5;
    const int half = lane >> 4;
    const int l16  = lane & 15;
    const int sBase = blockIdx.x * 32;
    const int vBase = (blockIdx.y * 4 + wave) * 32;
    const unsigned short* arow0 = A + (size_t)(sBase + l16) * HID;
    const unsigned short* arow1 = arow0 + (size_t)16 * HID;
    const unsigned short* brow0 = B + (size_t)(vBase + l16) * HID;
    const unsigned short* brow1 = brow0 + (size_t)16 * HID;
    v8f acc00 = {0.f,0.f,0.f,0.f,0.f,0.f,0.f,0.f};
    v8f acc01 = acc00, acc10 = acc00, acc11 = acc00;
#pragma unroll 2
    for (int k = 0; k < HID; k += 32) {
        BF16Tile a0, a1, b0, b1;
        const int aoff = k + 16 * half;
        const int boff = k + 16 * half;
        a0.q[0] = *reinterpret_cast<const i4x*>(arow0 + aoff);
        a0.q[1] = *reinterpret_cast<const i4x*>(arow0 + aoff + 8);
        a1.q[0] = *reinterpret_cast<const i4x*>(arow1 + aoff);
        a1.q[1] = *reinterpret_cast<const i4x*>(arow1 + aoff + 8);
        b0.q[0] = *reinterpret_cast<const i4x*>(brow0 + boff);
        b0.q[1] = *reinterpret_cast<const i4x*>(brow0 + boff + 8);
        b1.q[0] = *reinterpret_cast<const i4x*>(brow1 + boff);
        b1.q[1] = *reinterpret_cast<const i4x*>(brow1 + boff + 8);
        acc00 = wmma_bf16(a0, b0, acc00);
        acc01 = wmma_bf16(a0, b1, acc01);
        acc10 = wmma_bf16(a1, b0, acc10);
        acc11 = wmma_bf16(a1, b1, acc11);
    }
    const float bv0 = bias[vBase + l16];
    const float bv1 = bias[vBase + 16 + l16];
#pragma unroll
    for (int j = 0; j < 8; ++j) {
        const int r0 = sBase + j + 8 * half;
        const int r1 = r0 + 16;
        C[(size_t)r0 * NCH + vBase + l16]      = acc00[j] + bv0;
        C[(size_t)r0 * NCH + vBase + 16 + l16] = acc01[j] + bv1;
        C[(size_t)r1 * NCH + vBase + l16]      = acc10[j] + bv0;
        C[(size_t)r1 * NCH + vBase + 16 + l16] = acc11[j] + bv1;
    }
}

// ---------------- sequential recurrence: 16-WGP cluster ----------------
// Each of the 16 workgroups owns 64 rows of Waa resident in LDS (256 KB/WGP).
// h ping-pongs through global memory; steps separated by the cluster barrier.
// h is staged into LDS with the async global->LDS DMA path (ASYNCcnt).
__global__ void __launch_bounds__(256)
k_rnn(const float* __restrict__ Waa,      // [HID, HID] f32
      const float* __restrict__ xproj,    // [SEQ, HID] f32
      unsigned short* __restrict__ hsB,   // [SEQ, HID] bf16 (for output GEMM)
      float* __restrict__ hbuf,           // [2*HID] ping-pong
      float* __restrict__ hlast) {        // [HID]
    extern __shared__ float smem[];
    float* wsl = smem;                    // 64*HID Waa slice
    float* hS  = smem + 64 * HID;         // HID staged h
    const int wg   = blockIdx.x;          // 0..15 (cluster member)
    const int tid  = threadIdx.x;         // 0..255
    const int row  = tid >> 2;            // 0..63
    const int q    = tid & 3;             // quarter of the dot product
    const int rowg = wg * 64 + row;

    // Stage this WGP's 64 Waa rows into LDS (one-time; removes 256 GB of L2 traffic)
    const float* wsrc = Waa + (size_t)wg * 64 * HID;
    for (int i = tid; i < 64 * HID; i += 256) wsl[i] = wsrc[i];
    if (q == 0) { hbuf[rowg] = 0.f; hbuf[HID + rowg] = 0.f; }
    __threadfence();
    __syncthreads();
    __builtin_amdgcn_s_cluster_barrier();   // s_barrier_signal/-wait -3
    __threadfence();

    // byte offset of hS[tid*4] within the wave's LDS allocation
    const unsigned lds_off = (unsigned)(64 * HID * sizeof(float)) + (unsigned)tid * 16u;
    const float* wr = wsl + row * HID + q * 256;
    float lasth = 0.f;
    for (int t = 0; t < SEQ; ++t) {
        const float* hc = hbuf + ((t & 1) ? HID : 0);
        float*       hn = hbuf + ((t & 1) ? 0 : HID);
        // async DMA: 256 lanes x 16B = full 4KB h vector, global -> LDS (no VGPR bounce)
        {
            const float* gsrc = hc + tid * 4;
            asm volatile("global_load_async_to_lds_b128 %0, %1, off"
                         :: "v"(lds_off), "v"(gsrc) : "memory");
            asm volatile("s_wait_asynccnt 0x0" ::: "memory");
        }
        __syncthreads();
        if ((t & 7) == 0)
            __builtin_prefetch(xproj + (size_t)(t + 8) * HID + rowg, 0, 1);
        const float* hp = hS + q * 256;
        float sum = 0.f;
#pragma unroll 8
        for (int k = 0; k < 256; ++k) sum = fmaf(wr[k], hp[k], sum);
        sum += __shfl_xor(sum, 1, 32);
        sum += __shfl_xor(sum, 2, 32);
        if (q == 0) {
            float hv = tanhf(xproj[(size_t)t * HID + rowg] + sum);
            hn[rowg] = hv;
            hsB[(size_t)t * HID + rowg] = f32_to_bf16(hv);
            lasth = hv;
        }
        __threadfence();                      // release our h_new stores
        __syncthreads();
        __builtin_amdgcn_s_cluster_barrier(); // all 16 WGPs arrive
        __threadfence();                      // acquire others' h_new
    }
    if (q == 0) hlast[rowg] = lasth;
}

extern "C" void kernel_launch(void* const* d_in, const int* in_sizes, int n_in,
                              void* d_out, int out_size, void* d_ws, size_t ws_size,
                              hipStream_t stream) {
    (void)in_sizes; (void)n_in; (void)out_size; (void)ws_size;
    const int*   seq = (const int*)d_in[0];
    const float* emb = (const float*)d_in[1];
    const float* Wax = (const float*)d_in[2];
    const float* Waa = (const float*)d_in[3];
    const float* Wya = (const float*)d_in[4];
    const float* b_y = (const float*)d_in[5];
    float* out   = (float*)d_out;
    float* hlast = out + (size_t)SEQ * NCH;

    char* ws = (char*)d_ws;
    size_t off = 0;
    auto alloc = [&](size_t bytes) -> void* {
        void* p = ws + off;
        off = (off + bytes + 255) & ~(size_t)255;
        return p;
    };
    unsigned short* embB = (unsigned short*)alloc((size_t)SEQ * EMB * 2);
    unsigned short* WaxB = (unsigned short*)alloc((size_t)HID * EMB * 2);
    unsigned short* WyaB = (unsigned short*)alloc((size_t)NCH * HID * 2);
    float*          xprj = (float*)         alloc((size_t)SEQ * HID * 4);
    unsigned short* hsB  = (unsigned short*)alloc((size_t)SEQ * HID * 2);
    float*          hbuf = (float*)         alloc((size_t)2 * HID * 4);

    k_cvt_bf16<<<(HID * EMB + 255) / 256, 256, 0, stream>>>(Wax, WaxB, HID * EMB);
    k_cvt_bf16<<<(NCH * HID + 255) / 256, 256, 0, stream>>>(Wya, WyaB, NCH * HID);
    k_embed<<<(SEQ * EMB) / 256, 256, 0, stream>>>(seq, emb, embB);

    // 2048 s-blocks x 32 h-blocks of 32x32; 4 waves/block, 2x2 WMMA tiles per wave
    k_xproj<<<dim3(SEQ / 32, (HID / 32) / 4), 128, 0, stream>>>(embB, WaxB, xprj);

    // 16-workgroup cluster, 256 KB Waa slice + 4 KB h per WGP in dynamic LDS
    size_t rnn_lds = (size_t)(64 * HID + HID) * sizeof(float);
    k_rnn<<<16, 256, rnn_lds, stream>>>(Waa, xprj, hsB, hbuf, hlast);

    // 2048 s-blocks x 16 v-blocks of 32x32
    k_out<<<dim3(SEQ / 32, (NCH / 32) / 4), 128, 0, stream>>>(hsB, WyaB, b_y, out);
}
